// ELMo_32547262169715
// MI455X (gfx1250) — compile-verified, matrix operand order checked by reading, not run on hardware
//
#include <hip/hip_runtime.h>

// ---------------------------------------------------------------------------
// ELMo-style biLM for MI455X (gfx1250, wave32, WMMA).
// Precision: bf16 operands, f32 accumulate (v_wmma_f32_16x16x32_bf16).
// ---------------------------------------------------------------------------

typedef __attribute__((ext_vector_type(16))) __bf16          v16bf;
typedef __attribute__((ext_vector_type(8)))  float           v8f;
typedef __attribute__((ext_vector_type(8)))  unsigned short  v8us;
typedef __attribute__((ext_vector_type(16))) unsigned short  v16us;

#define B_   32
#define T_   256
#define V_   10000
#define E_   256
#define U_   512
#define G_   2048   // 4*U

static __device__ __forceinline__ unsigned short f2bf(float f) {
  unsigned int u = __float_as_uint(f);
  u += 0x7FFFu + ((u >> 16) & 1u);          // round-to-nearest-even
  return (unsigned short)(u >> 16);
}
static __device__ __forceinline__ float bf2f(unsigned short h) {
  return __uint_as_float(((unsigned int)h) << 16);
}
static __device__ __forceinline__ float sigf(float x) {
  return 1.0f / (1.0f + __expf(-x));
}

// Load a 16x32 bf16 A (or B^T) fragment from a row-major [rows x K] matrix.
// Lane L holds row (L&15); K chunks: [8*half, 8*half+8) then +16 (ISA 7.12.2).
// p must point at &M[row][k0 + 8*(lane>>4)], 16B aligned.
static __device__ __forceinline__ v16bf frag_ld(const unsigned short* p) {
  v8us lo = *(const v8us*)(p);
  v8us hi = *(const v8us*)(p + 16);
  v16us t;
#pragma unroll
  for (int i = 0; i < 8; ++i) { t[i] = lo[i]; t[i + 8] = hi[i]; }
  return __builtin_bit_cast(v16bf, t);
}

static __device__ __forceinline__ v8f wmma_bf16(v16bf a, v16bf b, v8f c) {
  return __builtin_amdgcn_wmma_f32_16x16x32_bf16(
      /*neg_a=*/false, a, /*neg_b=*/false, b,
      /*c_mod=*/(short)0, c, /*reuse_a=*/false, /*reuse_b=*/false);
}

// ---------------------------------------------------------------------------
// Kernel 1: embedding gather + f32 -> bf16
// ---------------------------------------------------------------------------
__global__ void embed_kernel(const int* __restrict__ seqs,
                             const float* __restrict__ E,
                             unsigned short* __restrict__ emb) {
  int i = blockIdx.x * blockDim.x + threadIdx.x;
  if (i < B_ * T_ * E_) {
    int e  = i & (E_ - 1);
    int bt = i >> 8;                 // E_ == 256
    int tok = seqs[bt];
    emb[i] = f2bf(E[tok * E_ + e]);
  }
}

// ---------------------------------------------------------------------------
// Kernel 2: pack [Wx;Wh] (row-major [K x N]) into transposed bf16 [N x K].
// Kh==0 -> plain transpose of Wx.
// ---------------------------------------------------------------------------
__global__ void pack_kernel(const float* __restrict__ Wx, int Kx,
                            const float* __restrict__ Wh, int Kh,
                            int N, unsigned short* __restrict__ outT) {
  int K = Kx + Kh;
  int i = blockIdx.x * blockDim.x + threadIdx.x;
  if (i < N * K) {
    int k = i % K;
    int n = i / K;
    float v = (k < Kx) ? Wx[k * N + n] : Wh[(k - Kx) * N + n];
    outT[i] = f2bf(v);
  }
}

// ---------------------------------------------------------------------------
// Shared GEMM tile pass for the scan: z[32 x 2048] = in[32 x K] @ W^T + bias.
// in: LDS, row-major, stride lda.  WT: global bf16 [2048 x K].
// 32 waves, 8 tiles each (2 M-tiles x 128 N-tiles). z written bf16 to LDS.
// Fragment loads are manually double-buffered so iteration k+1's global/LDS
// loads issue before WMMA k's wait; unrolling kept at 1 to avoid spills.
// ---------------------------------------------------------------------------
static __device__ __forceinline__ void gemm_tiles(
    const unsigned short* __restrict__ inb, int lda, int K,
    const unsigned short* __restrict__ WT,
    const float* __restrict__ bias,
    unsigned short* __restrict__ zb) {
  const int tid  = threadIdx.x;
  const int w    = tid >> 5;
  const int lane = tid & 31;
  const int r    = lane & 15;
  const int half = lane >> 4;
#pragma unroll 1
  for (int j = 0; j < 8; ++j) {
    const int tile = w * 8 + j;
    const int m0 = (tile & 1) * 16;
    const int n0 = (tile >> 1) * 16;
    const unsigned short* pa = inb + (m0 + r) * lda + 8 * half;
    const unsigned short* pb = WT + (size_t)(n0 + r) * K + 8 * half;
    v8f acc = {0.f, 0.f, 0.f, 0.f, 0.f, 0.f, 0.f, 0.f};
    v16bf a = frag_ld(pa);
    v16bf b = frag_ld(pb);
#pragma unroll 1
    for (int k0 = 32; k0 < K; k0 += 32) {
      v16bf an = frag_ld(pa + k0);               // prefetch next pair before
      v16bf bn = frag_ld(pb + k0);               // the current WMMA's wait
      __builtin_prefetch(pb + k0 + 224, 0, 3);
      acc = wmma_bf16(a, b, acc);
      a = an;
      b = bn;
    }
    acc = wmma_bf16(a, b, acc);
    const float bv = bias[n0 + r];
    unsigned short* zp = zb + n0 + r;
#pragma unroll
    for (int rr = 0; rr < 8; ++rr) {
      int m = m0 + rr + 8 * half;                // C/D layout: VGPR rr -> row
      zp[m * G_] = f2bf(acc[rr] + bv);
    }
  }
}

// ---------------------------------------------------------------------------
// Kernel 3: persistent 2-layer LSTM scan. blockIdx.x: 0 = forward, 1 = backward.
// LDS: in0 [32x768] = [x_t | h0_prev], in1 [32x1024] = [h0_t | h1_prev],
//      zb [32x2048] bf16 gate pre-activations.  (240 KB of 320 KB WGP LDS)
// c state in registers (16 elems/thread/layer); carried h lives in LDS (bf16).
// ---------------------------------------------------------------------------
__global__ __launch_bounds__(1024) void lstm_scan_kernel(
    const int* __restrict__ seqs,
    const unsigned short* __restrict__ emb,
    const unsigned short* __restrict__ W0T_f,
    const unsigned short* __restrict__ W1T_f,
    const float* __restrict__ b0_f, const float* __restrict__ b1_f,
    const unsigned short* __restrict__ W0T_b,
    const unsigned short* __restrict__ W1T_b,
    const float* __restrict__ b0_b, const float* __restrict__ b1_b,
    float* __restrict__ f_out, float* __restrict__ b_out,
    unsigned short* __restrict__ obuf) {
  extern __shared__ unsigned short smem[];
  unsigned short* in0 = smem;                     // 32*768
  unsigned short* in1 = smem + B_ * 768;          // 32*1024
  unsigned short* zb  = in1 + B_ * 1024;          // 32*2048

  const int dir = blockIdx.x;
  const unsigned short* W0T = dir ? W0T_b : W0T_f;
  const unsigned short* W1T = dir ? W1T_b : W1T_f;
  const float* bias0 = dir ? b0_b : b0_f;
  const float* bias1 = dir ? b1_b : b1_f;

  const int tid = threadIdx.x;

  float c0[16], c1[16];
#pragma unroll
  for (int i = 0; i < 16; ++i) { c0[i] = 0.f; c1[i] = 0.f; }
  // Zero the h_prev regions of the LDS input buffers.
#pragma unroll
  for (int i = 0; i < 16; ++i) {
    int e = i * 1024 + tid;
    int b = e >> 9, u = e & (U_ - 1);
    in0[b * 768 + 256 + u]  = 0;
    in1[b * 1024 + 512 + u] = 0;
  }
  __syncthreads();

#pragma unroll 1
  for (int t = 0; t < 255; ++t) {
    const int tok = dir ? (255 - t) : t;         // bwd consumes emb[:,1:] reversed
    // ---- stage x_t into in0[:, 0:256] (one b128 per thread) ----
    {
      int b  = tid >> 5;
      int c8 = (tid & 31) * 8;
      *(v8us*)(in0 + b * 768 + c8) =
          *(const v8us*)(emb + ((size_t)(b * T_ + tok)) * E_ + c8);
    }
    __syncthreads();

    // ---- layer 0: z = [x | h0_prev] @ [Wx0;Wh0]^T + b0 ----
    gemm_tiles(in0, 768, 768, W0T, bias0, zb);
    __syncthreads();

    // ---- layer 0 gates ----
#pragma unroll 1
    for (int i = 0; i < 16; ++i) {
      int e = i * 1024 + tid;
      int b = e >> 9, u = e & (U_ - 1);
      float zi = bf2f(zb[b * G_ + u]);
      float zf = bf2f(zb[b * G_ + U_ + u]);
      float zg = bf2f(zb[b * G_ + 2 * U_ + u]);
      float zo = bf2f(zb[b * G_ + 3 * U_ + u]);
      float hp = bf2f(in0[b * 768 + 256 + u]);   // carried h0 (bf16)
      float cn = sigf(zf) * c0[i] + sigf(zi) * tanhf(zg);
      float hn = sigf(zo) * tanhf(cn);
      bool  m  = (seqs[b * T_ + tok] != 0);
      float h  = m ? hn : hp;
      c0[i]    = m ? cn : c0[i];
      unsigned short hb = f2bf(h);
      in0[b * 768 + 256 + u] = hb;               // h0_prev for next step
      in1[b * 1024 + u]      = hb;               // layer-1 input this step
    }
    __syncthreads();

    // ---- layer 1: z = [h0 | h1_prev] @ [Wx1;Wh1]^T + b1 ----
    gemm_tiles(in1, 1024, 1024, W1T, bias1, zb);
    __syncthreads();

    // ---- layer 1 gates + outputs ----
#pragma unroll 1
    for (int i = 0; i < 16; ++i) {
      int e = i * 1024 + tid;
      int b = e >> 9, u = e & (U_ - 1);
      float zi = bf2f(zb[b * G_ + u]);
      float zf = bf2f(zb[b * G_ + U_ + u]);
      float zg = bf2f(zb[b * G_ + 2 * U_ + u]);
      float zo = bf2f(zb[b * G_ + 3 * U_ + u]);
      float hp = bf2f(in1[b * 1024 + 512 + u]);  // carried h1 (bf16)
      float cn = sigf(zf) * c1[i] + sigf(zi) * tanhf(zg);
      float hn = sigf(zo) * tanhf(cn);
      bool  m  = (seqs[b * T_ + tok] != 0);
      float h  = m ? hn : hp;
      c1[i]    = m ? cn : c1[i];
      unsigned short hb = f2bf(h);
      in1[b * 1024 + 512 + u] = hb;              // h1_prev for next step
      if (dir == 0) {
        f_out[((size_t)(b * 255 + t)) * U_ + u] = h;
        if (t < 254) obuf[((size_t)(b * 254 + t)) * 1024 + u] = hb;       // o[:,t,0:512]
      } else {
        b_out[((size_t)(b * 255 + (254 - t))) * U_ + u] = h;              // b_reverse
        if (t < 254) obuf[((size_t)(b * 254 + (253 - t))) * 1024 + 512 + u] = hb;
      }
    }
    __syncthreads();
  }
}

// ---------------------------------------------------------------------------
// Kernel 4: projection GEMM  logits[8128 x 10000] = o[8128 x 1024] @ Wp + bp.
// A = obuf bf16 row-major; B = WpT bf16 [10000 x 1024] (transposed). One wave
// per 16x16 tile; block = 8 waves = 2 M-tiles x 4 N-tiles. Double-buffered
// fragment pipeline as in the scan.
// ---------------------------------------------------------------------------
__global__ __launch_bounds__(256) void proj_kernel(
    const unsigned short* __restrict__ obuf,
    const unsigned short* __restrict__ WpT,
    const float* __restrict__ bp,
    float* __restrict__ out) {
  const int w    = threadIdx.x >> 5;
  const int lane = threadIdx.x & 31;
  const int mt = blockIdx.y * 2 + (w >> 2);      // 0..507
  const int nt = blockIdx.x * 4 + (w & 3);       // 0..627 (625 valid)
  if (nt >= V_ / 16) return;
  const int m0 = mt * 16, n0 = nt * 16;
  const int r = lane & 15, half = lane >> 4;
  const unsigned short* pa = obuf + (size_t)(m0 + r) * 1024 + 8 * half;
  const unsigned short* pb = WpT  + (size_t)(n0 + r) * 1024 + 8 * half;
  v8f acc = {0.f, 0.f, 0.f, 0.f, 0.f, 0.f, 0.f, 0.f};
  v16bf a = frag_ld(pa);
  v16bf b = frag_ld(pb);
#pragma unroll 1
  for (int k0 = 32; k0 < 1024; k0 += 32) {
    v16bf an = frag_ld(pa + k0);
    v16bf bn = frag_ld(pb + k0);
    __builtin_prefetch(pb + k0 + 224, 0, 3);
    acc = wmma_bf16(a, b, acc);
    a = an;
    b = bn;
  }
  acc = wmma_bf16(a, b, acc);
  const float bv = bp[n0 + r];
#pragma unroll
  for (int rr = 0; rr < 8; ++rr) {
    int m = m0 + rr + 8 * half;
    out[(size_t)m * V_ + n0 + r] = acc[rr] + bv;
  }
}

// ---------------------------------------------------------------------------
// Host launcher
// ---------------------------------------------------------------------------
extern "C" void kernel_launch(void* const* d_in, const int* in_sizes, int n_in,
                              void* d_out, int out_size, void* d_ws, size_t ws_size,
                              hipStream_t stream) {
  const int*   seqs  = (const int*)d_in[0];
  const float* E     = (const float*)d_in[1];
  const float* f_Wx0 = (const float*)d_in[2];
  const float* f_Wh0 = (const float*)d_in[3];
  const float* f_b0  = (const float*)d_in[4];
  const float* f_Wx1 = (const float*)d_in[5];
  const float* f_Wh1 = (const float*)d_in[6];
  const float* f_b1  = (const float*)d_in[7];
  const float* b_Wx0 = (const float*)d_in[8];
  const float* b_Wh0 = (const float*)d_in[9];
  const float* b_b0  = (const float*)d_in[10];
  const float* b_Wx1 = (const float*)d_in[11];
  const float* b_Wh1 = (const float*)d_in[12];
  const float* b_b1  = (const float*)d_in[13];
  const float* Wp    = (const float*)d_in[14];
  const float* bp    = (const float*)d_in[15];

  // Workspace carve-up (bf16 buffers).
  unsigned short* emb  = (unsigned short*)d_ws;          // 32*256*256
  unsigned short* W0Tf = emb  + (size_t)B_ * T_ * E_;    // 2048*768
  unsigned short* W1Tf = W0Tf + (size_t)G_ * 768;        // 2048*1024
  unsigned short* W0Tb = W1Tf + (size_t)G_ * 1024;       // 2048*768
  unsigned short* W1Tb = W0Tb + (size_t)G_ * 768;        // 2048*1024
  unsigned short* WpT  = W1Tb + (size_t)G_ * 1024;       // 10000*1024
  unsigned short* obuf = WpT  + (size_t)V_ * 1024;       // 32*254*1024

  // d_out layout: word_logits | f | b_reverse  (all f32)
  float* logits = (float*)d_out;
  float* f_out  = logits + (size_t)B_ * 254 * V_;
  float* b_out  = f_out  + (size_t)B_ * 255 * U_;

  // 1) embedding gather + bf16 cast
  {
    int n = B_ * T_ * E_;
    embed_kernel<<<(n + 255) / 256, 256, 0, stream>>>(seqs, E, emb);
  }
  // 2) weight packing: [Wx;Wh] -> bf16 transposed [N x K]
  {
    int n;
    n = G_ * 768;
    pack_kernel<<<(n + 255) / 256, 256, 0, stream>>>(f_Wx0, 256, f_Wh0, 512, G_, W0Tf);
    pack_kernel<<<(n + 255) / 256, 256, 0, stream>>>(b_Wx0, 256, b_Wh0, 512, G_, W0Tb);
    n = G_ * 1024;
    pack_kernel<<<(n + 255) / 256, 256, 0, stream>>>(f_Wx1, 512, f_Wh1, 512, G_, W1Tf);
    pack_kernel<<<(n + 255) / 256, 256, 0, stream>>>(b_Wx1, 512, b_Wh1, 512, G_, W1Tb);
    n = V_ * 1024;
    pack_kernel<<<(n + 255) / 256, 256, 0, stream>>>(Wp, 1024, Wp, 0, V_, WpT);
  }
  // 3) persistent biLSTM scan: block 0 = fwd, block 1 = bwd
  {
    size_t lds = (size_t)(B_ * 768 + B_ * 1024 + B_ * G_) * sizeof(unsigned short); // 240 KB
    lstm_scan_kernel<<<2, 1024, lds, stream>>>(
        seqs, emb, W0Tf, W1Tf, f_b0, f_b1, W0Tb, W1Tb, b_b0, b_b1,
        f_out, b_out, obuf);
  }
  // 4) projection GEMM
  {
    dim3 grid((V_ / 16 + 3) / 4, (B_ * 254 / 16) / 2);   // (157, 254)
    proj_kernel<<<grid, 256, 0, stream>>>(obuf, WpT, bp, logits);
  }
}
